// TransfusionTransformer_19739669692554
// MI455X (gfx1250) — matrix operand order repaired and checked by reading, not compiled, not run on hardware
//
#include <hip/hip_runtime.h>
#include <hip/hip_bf16.h>

#define DIMD   1024
#define SEQ    1024
#define NHEAD  16
#define HDIM   64
#define FFDIM  2816
#define VOCAB  32000
#define BOI_TOK 31998
#define EOI_TOK 31999

typedef __attribute__((ext_vector_type(16))) __bf16 v16bf;
typedef __attribute__((ext_vector_type(8)))  float  v8f;

// ---------------- bf16 helpers (RNE) ----------------
static __device__ __forceinline__ __bf16 f2bf(float f) {
    union { float f; unsigned u; } a; a.f = f;
    unsigned r = a.u + 0x7FFFu + ((a.u >> 16) & 1u);
    union { unsigned short s; __bf16 b; } o; o.s = (unsigned short)(r >> 16);
    return o.b;
}
static __device__ __forceinline__ float bf2f(__bf16 b) {
    union { unsigned short s; __bf16 b; } i; i.b = b;
    union { unsigned u; float f; } o; o.u = ((unsigned)i.s) << 16;
    return o.f;
}

// ---------------- WMMA fragment loaders (contiguous k = 16h + s, same order for A and B) ----------------
// A 16x32 bf16: lane<16 row=lane, K=c0..c0+15 contiguous with c0=0;
// lane>=16 same row, c0=16. Caller passes p = &row[k + c0]; reads p[0..15].
static __device__ __forceinline__ v16bf loadA16(const __bf16* p) {
    union { v16bf v; uint4 q[2]; } u;
    u.q[0] = *(const uint4*)(p);
    u.q[1] = *(const uint4*)(p + 8);
    return u.v;
}
// B 32x16 bf16: lane n<16 col=n rows K=0..15 contiguous; lane>=16 rows 16..31.
// Caller passes p = 16 contiguous bf16.
static __device__ __forceinline__ v16bf loadB16(const __bf16* p) {
    union { v16bf v; uint4 q[2]; } u;
    u.q[0] = *(const uint4*)(p);
    u.q[1] = *(const uint4*)(p + 8);
    return u.v;
}
static __device__ __forceinline__ v8f wmma_bf16(v16bf a, v16bf b, v8f c) {
    return __builtin_amdgcn_wmma_f32_16x16x32_bf16(false, a, false, b, (short)0, c, false, false);
}

// ---------------- weight f32 -> bf16 streaming conversion ----------------
// One pass per weight tensor per launch; keeps conversion VALU off the GEMM
// critical path. 8 elements/thread: 2x float4 in, 1x uint4 (8 bf16) out.
__global__ __launch_bounds__(256) void cvtw_k(
    const float* __restrict__ src, __bf16* __restrict__ dst)
{
    const size_t i = ((size_t)blockIdx.x * 256 + threadIdx.x) * 8;
    float4 f0 = *(const float4*)(src + i);
    float4 f1 = *(const float4*)(src + i + 4);
    union { __bf16 b[8]; uint4 q; } u;
    u.b[0] = f2bf(f0.x); u.b[1] = f2bf(f0.y);
    u.b[2] = f2bf(f0.z); u.b[3] = f2bf(f0.w);
    u.b[4] = f2bf(f1.x); u.b[5] = f2bf(f1.y);
    u.b[6] = f2bf(f1.z); u.b[7] = f2bf(f1.w);
    *(uint4*)(dst + i) = u.q;
}

// ---------------- GEMM: Y[M,N] = X[M,K](bf16) @ WB[N,K](bf16)^T ----------------
// 256 threads = 8 waves; wave tile 32M x 64N (2 A-frags x 4 B-frags = 8 WMMA
// per K-step of 32, 12 b128 loads). Workgroup tile 64M x 256N.
enum { EPI_F32 = 0, EPI_F32_RESID = 1, EPI_BF16_QKV = 2, EPI_BF16_VT = 3 };

template <int EPI>
__global__ __launch_bounds__(256) void gemm_xwT(
    const __bf16* __restrict__ X, const __bf16* __restrict__ WB,
    void* __restrict__ Yv, const float* __restrict__ resid,
    int M, int N, int K)
{
    const int lane = threadIdx.x & 31;
    const int wv   = threadIdx.x >> 5;            // 8 waves
    const int mblk = blockIdx.y * 64  + (wv >> 2) * 32;
    const int nblk = blockIdx.x * 256 + (wv & 3) * 64;
    const int hi   = lane >> 4;
    const int l15  = lane & 15;

    v8f acc[2][4] = {};

    const __bf16* xp0 = X + (size_t)(mblk + l15) * K + hi * 16;
    const __bf16* xp1 = xp0 + (size_t)16 * K;
    const __bf16* wp0 = WB + (size_t)(nblk + l15) * K + hi * 16;
    const __bf16* wp1 = wp0 + (size_t)16 * K;
    const __bf16* wp2 = wp0 + (size_t)32 * K;
    const __bf16* wp3 = wp0 + (size_t)48 * K;

    for (int k = 0; k < K; k += 32) {
        if (k + 32 < K) {                         // pull next weight K-slab toward L0/L2
            __builtin_prefetch(wp0 + k + 32, 0, 1);
            __builtin_prefetch(wp2 + k + 32, 0, 1);
        }
        v16bf a0 = loadA16(xp0 + k);
        v16bf a1 = loadA16(xp1 + k);
        v16bf b0 = loadB16(wp0 + k);
        v16bf b1 = loadB16(wp1 + k);
        v16bf b2 = loadB16(wp2 + k);
        v16bf b3 = loadB16(wp3 + k);
        acc[0][0] = wmma_bf16(a0, b0, acc[0][0]);
        acc[0][1] = wmma_bf16(a0, b1, acc[0][1]);
        acc[0][2] = wmma_bf16(a0, b2, acc[0][2]);
        acc[0][3] = wmma_bf16(a0, b3, acc[0][3]);
        acc[1][0] = wmma_bf16(a1, b0, acc[1][0]);
        acc[1][1] = wmma_bf16(a1, b1, acc[1][1]);
        acc[1][2] = wmma_bf16(a1, b2, acc[1][2]);
        acc[1][3] = wmma_bf16(a1, b3, acc[1][3]);
    }

#pragma unroll
    for (int i = 0; i < 2; i++)
#pragma unroll
        for (int j = 0; j < 4; j++)
#pragma unroll
            for (int e = 0; e < 8; e++) {
                int row = mblk + i * 16 + e + hi * 8;   // C layout: VGPR e -> M=e (+8 upper half)
                int col = nblk + j * 16 + l15;
                float val = acc[i][j][e];
                if constexpr (EPI == EPI_F32) {
                    ((float*)Yv)[(size_t)row * N + col] = val;
                } else if constexpr (EPI == EPI_F32_RESID) {
                    ((float*)Yv)[(size_t)row * N + col] =
                        resid[(size_t)row * N + col] + val;
                } else if constexpr (EPI == EPI_BF16_QKV) {     // -> [H][L][64]
                    int hh = col >> 6, d = col & 63;
                    ((__bf16*)Yv)[((size_t)hh * M + row) * 64 + d] = f2bf(val);
                } else {                                         // EPI_BF16_VT -> [H][64][L]
                    ((__bf16*)Yv)[(size_t)col * M + row] = f2bf(val);
                }
            }
}

// ---------------- Flash attention: 1 wave per (head, 16-query tile) ----------------
__global__ __launch_bounds__(32) void attn_fwd(
    const __bf16* __restrict__ Q,   // [H][L][64]
    const __bf16* __restrict__ Kh,  // [H][L][64]
    const __bf16* __restrict__ Vt,  // [H][64][L]
    const int*    __restrict__ sid, // span id per position (0 = not in span)
    __bf16*       __restrict__ out) // [L][D]
{
    __shared__ __bf16 psh[16 * 32];                    // P staging for layout transpose
    const int bid   = blockIdx.x;
    const int h     = bid >> 6;                        // L/16 = 64 q-tiles
    const int qbase = (bid & 63) << 4;
    const int lane  = threadIdx.x;
    const int hi    = lane >> 4;
    const int l15   = lane & 15;
    const float scale = 0.125f;                        // 1/sqrt(64)

    const __bf16* Qb = Q + ((size_t)h * SEQ + qbase) * HDIM;
    v16bf aq0 = loadA16(Qb + (size_t)l15 * HDIM + hi * 16);        // head-dim 0..31
    v16bf aq1 = loadA16(Qb + (size_t)l15 * HDIM + 32 + hi * 16);   // head-dim 32..63

    float m_i[8], l_i[8], fsc[8];
    int   sidq[8], qrow[8];
    v8f   o[4] = {};
#pragma unroll
    for (int e = 0; e < 8; e++) {
        m_i[e] = -1e30f; l_i[e] = 0.f;
        qrow[e] = qbase + e + hi * 8;
        sidq[e] = sid[qrow[e]];
    }

    for (int kb = 0; kb < SEQ; kb += 32) {
        const __bf16* Kb = Kh + ((size_t)h * SEQ + kb) * HDIM;
        v8f s0 = {}, s1 = {};
        {   // scores for keys kb..kb+15 (s0) and kb+16..kb+31 (s1)
            v16bf bk00 = loadB16(Kb + (size_t)l15 * HDIM + hi * 16);
            v16bf bk01 = loadB16(Kb + (size_t)l15 * HDIM + 32 + hi * 16);
            v16bf bk10 = loadB16(Kb + (size_t)(16 + l15) * HDIM + hi * 16);
            v16bf bk11 = loadB16(Kb + (size_t)(16 + l15) * HDIM + 32 + hi * 16);
            s0 = wmma_bf16(aq0, bk00, s0); s0 = wmma_bf16(aq1, bk01, s0);
            s1 = wmma_bf16(aq0, bk10, s1); s1 = wmma_bf16(aq1, bk11, s1);
        }
        const int k0 = kb + l15, k1 = kb + 16 + l15;
        const int sk0 = sid[k0], sk1 = sid[k1];

#pragma unroll
        for (int e = 0; e < 8; e++) {
            float v0 = s0[e] * scale, v1 = s1[e] * scale;
            bool a0 = (qrow[e] >= k0) || (sidq[e] > 0 && sidq[e] == sk0);
            bool a1 = (qrow[e] >= k1) || (sidq[e] > 0 && sidq[e] == sk1);
            if (!a0) v0 = -1e30f;
            if (!a1) v1 = -1e30f;
            float mx = fmaxf(v0, v1);
#pragma unroll
            for (int off = 1; off < 16; off <<= 1)         // row-max across 16-lane half
                mx = fmaxf(mx, __shfl_xor(mx, off, 32));
            float nm = fmaxf(m_i[e], mx);
            float p0 = __expf(v0 - nm), p1 = __expf(v1 - nm);
            float sm = p0 + p1;
#pragma unroll
            for (int off = 1; off < 16; off <<= 1)
                sm += __shfl_xor(sm, off, 32);
            float f = __expf(m_i[e] - nm);
            l_i[e] = l_i[e] * f + sm;
            m_i[e] = nm;
            fsc[e] = f;
            psh[(e + hi * 8) * 32 + l15]      = f2bf(p0);  // C-layout -> row-major LDS
            psh[(e + hi * 8) * 32 + 16 + l15] = f2bf(p1);
        }
#pragma unroll
        for (int t = 0; t < 4; t++)
#pragma unroll
            for (int e = 0; e < 8; e++) o[t][e] *= fsc[e];

        asm volatile("s_wait_dscnt 0" ::: "memory");       // LDS stores visible before reload
        v16bf pa = loadA16(psh + l15 * 32 + hi * 16);      // re-read P in A-frag layout

        const __bf16* Vb = Vt + (size_t)h * HDIM * SEQ + kb + hi * 16;
#pragma unroll
        for (int t = 0; t < 4; t++) {                      // P(16x32) @ V(32x64)
            v16bf bv = loadB16(Vb + (size_t)(t * 16 + l15) * SEQ);
            o[t] = wmma_bf16(pa, bv, o[t]);
        }
    }

#pragma unroll
    for (int e = 0; e < 8; e++) fsc[e] = 1.f / l_i[e];
#pragma unroll
    for (int t = 0; t < 4; t++)
#pragma unroll
        for (int e = 0; e < 8; e++) {
            int row = qbase + e + hi * 8;
            out[(size_t)row * DIMD + h * 64 + t * 16 + l15] = f2bf(o[t][e] * fsc[e]);
        }
}

// ---------------- RMSNorm: f32 in -> bf16*w out ----------------
__global__ __launch_bounds__(256) void rmsnorm_k(
    const float* __restrict__ x, const float* __restrict__ w, __bf16* __restrict__ y)
{
    const int row = blockIdx.x, tid = threadIdx.x;
    __shared__ float red[256];
    const float* xr = x + (size_t)row * DIMD;
    float s = 0.f;
    for (int c = tid; c < DIMD; c += 256) { float v = xr[c]; s += v * v; }
    red[tid] = s; __syncthreads();
    for (int o = 128; o > 0; o >>= 1) { if (tid < o) red[tid] += red[tid + o]; __syncthreads(); }
    const float inv = rsqrtf(red[0] * (1.f / DIMD) + 1e-5f);
    __bf16* yr = y + (size_t)row * DIMD;
    for (int c = tid; c < DIMD; c += 256) yr[c] = f2bf(xr[c] * inv * w[c]);
}

// ---------------- RoPE in-place on bf16 [H][L][64] ----------------
__global__ __launch_bounds__(32) void rope_k(__bf16* __restrict__ Q, __bf16* __restrict__ Kh)
{
    const int bid = blockIdx.x;
    const int h = bid >> 10, l = bid & (SEQ - 1);
    const int d = threadIdx.x;                       // pair index 0..31
    const float inv = __expf(-(2.f * d / (float)HDIM) * __logf(500000.f));
    float s, c;
    __sincosf((float)l * inv, &s, &c);
    const size_t base = ((size_t)h * SEQ + l) * HDIM + 2 * d;
    float qe = bf2f(Q[base]), qo = bf2f(Q[base + 1]);
    Q[base] = f2bf(qe * c - qo * s); Q[base + 1] = f2bf(qe * s + qo * c);
    float ke = bf2f(Kh[base]), ko = bf2f(Kh[base + 1]);
    Kh[base] = f2bf(ke * c - ko * s); Kh[base + 1] = f2bf(ke * s + ko * c);
}

// ---------------- misc elementwise ----------------
__global__ __launch_bounds__(256) void embed_k(
    const int* __restrict__ tok, const float* __restrict__ emb, float* __restrict__ h)
{
    const int l = blockIdx.x, t = tok[l];
    const float* e = emb + (size_t)t * DIMD;
    float* hr = h + (size_t)l * DIMD;
    for (int c = threadIdx.x; c < DIMD; c += 256) hr[c] = e[c];
}

__global__ void sid_k(const int* __restrict__ tok, int* __restrict__ sid)
{
    if (threadIdx.x == 0 && blockIdx.x == 0) {
        int boi = 0, eoi = 0;
        for (int i = 0; i < SEQ; i++) {
            int t = tok[i];
            if (t == BOI_TOK) boi++;
            int eoi_before = eoi;                 // exclusive cumsum of EOI
            sid[i] = (boi > eoi_before) ? boi : 0;
            if (t == EOI_TOK) eoi++;
        }
    }
}

__global__ __launch_bounds__(256) void silu_k(
    const float* __restrict__ a, const float* __restrict__ b, __bf16* __restrict__ g)
{
    const size_t i = (size_t)blockIdx.x * 256 + threadIdx.x;
    float x = a[i];
    float s = x / (1.f + __expf(-x));
    g[i] = f2bf(s * b[i]);
}

// ---------------- driver ----------------
extern "C" void kernel_launch(void* const* d_in, const int* in_sizes, int n_in,
                              void* d_out, int out_size, void* d_ws, size_t ws_size,
                              hipStream_t stream)
{
    (void)in_sizes; (void)n_in; (void)out_size; (void)ws_size;
    const int*   tokens  = (const int*)  d_in[0];
    const float* tok_emb = (const float*)d_in[1];
    const float* wq      = (const float*)d_in[2];
    const float* wk      = (const float*)d_in[3];
    const float* wv      = (const float*)d_in[4];
    const float* wo      = (const float*)d_in[5];
    const float* w1      = (const float*)d_in[6];
    const float* w2      = (const float*)d_in[7];
    const float* w3      = (const float*)d_in[8];
    const float* anw     = (const float*)d_in[9];
    const float* fnw     = (const float*)d_in[10];
    const float* finw    = (const float*)d_in[11];
    const float* w_out   = (const float*)d_in[12];

    char* wsp = (char*)d_ws;
    auto alloc = [&](size_t b) -> void* {
        void* r = (void*)wsp;
        wsp += (b + 255) & ~(size_t)255;
        return r;
    };
    float*  h    = (float*) alloc(sizeof(float) * SEQ * DIMD);       // f32 residual
    __bf16* xn   = (__bf16*)alloc(2ull * SEQ * DIMD);                // normed acts (bf16)
    __bf16* qh   = (__bf16*)alloc(2ull * NHEAD * SEQ * HDIM);        // Q [H][L][64]
    __bf16* kh   = (__bf16*)alloc(2ull * NHEAD * SEQ * HDIM);        // K [H][L][64]
    __bf16* vt   = (__bf16*)alloc(2ull * NHEAD * SEQ * HDIM);        // V [H][64][L]
    __bf16* att  = (__bf16*)alloc(2ull * SEQ * DIMD);                // attn out [L][D]
    float*  f1   = (float*) alloc(sizeof(float) * SEQ * FFDIM);
    float*  f3   = (float*) alloc(sizeof(float) * SEQ * FFDIM);
    __bf16* gact = (__bf16*)alloc(2ull * SEQ * FFDIM);
    __bf16* wb   = (__bf16*)alloc(2ull * VOCAB * DIMD);              // bf16 weight staging
    int*    sid  = (int*)   alloc(sizeof(int) * SEQ);

    const int nDD = DIMD * DIMD;                  // per-layer proj weight elems
    const int nFD = FFDIM * DIMD;                 // per-layer FFN weight elems
    const int nVD = VOCAB * DIMD;                 // output head weight elems

    embed_k<<<SEQ, 256, 0, stream>>>(tokens, tok_emb, h);
    sid_k<<<1, 1, 0, stream>>>(tokens, sid);

    const dim3 gD(DIMD / 256, SEQ / 64);          // N=1024 GEMMs
    const dim3 gF(FFDIM / 256, SEQ / 64);         // N=2816 GEMMs
    for (int l = 0; l < 2; l++) {
        rmsnorm_k<<<SEQ, 256, 0, stream>>>(h, anw + (size_t)l * DIMD, xn);

        cvtw_k<<<nDD / 2048, 256, 0, stream>>>(wq + (size_t)l * nDD, wb);
        gemm_xwT<EPI_BF16_QKV><<<gD, 256, 0, stream>>>(xn, wb, qh, nullptr, SEQ, DIMD, DIMD);
        cvtw_k<<<nDD / 2048, 256, 0, stream>>>(wk + (size_t)l * nDD, wb);
        gemm_xwT<EPI_BF16_QKV><<<gD, 256, 0, stream>>>(xn, wb, kh, nullptr, SEQ, DIMD, DIMD);
        cvtw_k<<<nDD / 2048, 256, 0, stream>>>(wv + (size_t)l * nDD, wb);
        gemm_xwT<EPI_BF16_VT><<<gD, 256, 0, stream>>>(xn, wb, vt, nullptr, SEQ, DIMD, DIMD);

        rope_k<<<NHEAD * SEQ, 32, 0, stream>>>(qh, kh);
        attn_fwd<<<NHEAD * (SEQ / 16), 32, 0, stream>>>(qh, kh, vt, sid, att);

        cvtw_k<<<nDD / 2048, 256, 0, stream>>>(wo + (size_t)l * nDD, wb);
        gemm_xwT<EPI_F32_RESID><<<gD, 256, 0, stream>>>(att, wb, h, h, SEQ, DIMD, DIMD);

        rmsnorm_k<<<SEQ, 256, 0, stream>>>(h, fnw + (size_t)l * DIMD, xn);

        cvtw_k<<<nFD / 2048, 256, 0, stream>>>(w1 + (size_t)l * nFD, wb);
        gemm_xwT<EPI_F32><<<gF, 256, 0, stream>>>(xn, wb, f1, nullptr, SEQ, FFDIM, DIMD);
        cvtw_k<<<nFD / 2048, 256, 0, stream>>>(w3 + (size_t)l * nFD, wb);
        gemm_xwT<EPI_F32><<<gF, 256, 0, stream>>>(xn, wb, f3, nullptr, SEQ, FFDIM, DIMD);

        silu_k<<<(SEQ * FFDIM) / 256, 256, 0, stream>>>(f1, f3, gact);

        cvtw_k<<<nFD / 2048, 256, 0, stream>>>(w2 + (size_t)l * nFD, wb);
        gemm_xwT<EPI_F32_RESID><<<gD, 256, 0, stream>>>(gact, wb, h, h, SEQ, DIMD, FFDIM);
    }

    rmsnorm_k<<<SEQ, 256, 0, stream>>>(h, finw, xn);
    cvtw_k<<<nVD / 2048, 256, 0, stream>>>(w_out, wb);
    const dim3 gO(VOCAB / 256, SEQ / 64);
    gemm_xwT<EPI_F32><<<gO, 256, 0, stream>>>(xn, wb, (float*)d_out, nullptr, SEQ, VOCAB, DIMD);
}
